// StackedLayerNormGRU_15006615734216
// MI455X (gfx1250) — compile-verified, matrix operand order checked by reading, not run on hardware
//
#include <hip/hip_runtime.h>

// ---------------------------------------------------------------------------
// Stacked LayerNorm-GRU, persistent single-cluster kernel for MI455X (gfx1250)
//   B=64, C_IN=256, H=512, T=1024, 3H=1536
// 16 workgroups (intended as one workgroup-cluster, one WGP each), 384 thr/WG.
// Each WG owns 32 hidden columns => 96 gate rows {r,i,n chunks}.
// GEMMs in bf16 via v_wmma_f32_16x16x32_bf16; weights pre-packed to the
// per-lane B-fragment layout; A operands staged in LDS.
// Per-step sync: CDNA5 cluster barrier (s_barrier_signal/-wait -3) backed by a
// monotonic global-atomic barrier for plain (non-cluster) launches.
// ---------------------------------------------------------------------------

#define B_    64
#define CIN_  256
#define H_    512
#define T_    1024
#define G3_   1536
#define NWG_  16
#define COLS_ 96      // gate columns per WG (3 chunks of 32)
#define HC_   32      // hidden columns per WG
#define NTHR_ 384     // 12 waves of 32
#define EPS_  1e-5f

typedef __attribute__((ext_vector_type(16))) __bf16 v16bf;
typedef __attribute__((ext_vector_type(8)))  float  v8f;

union FragU { uint4 q[2]; v16bf v; };

// 16-bit WMMA operand layout (wave32): lane L -> row (L&15); K chunks
// [8*(L>>4), +8) and [16+8*(L>>4), +8) within each 32-wide K slab.
__device__ __forceinline__ v16bf load_frag(const __bf16* p, int half) {
  FragU u;
  u.q[0] = *(const uint4*)(p + 8 * half);
  u.q[1] = *(const uint4*)(p + 16 + 8 * half);
  return u.v;
}

__device__ __forceinline__ v8f wmma_bf16(v16bf a, v16bf b, v8f c) {
  // (neg_a, A, neg_b, B, c_mod, C, reuse_a, reuse_b)
  return __builtin_amdgcn_wmma_f32_16x16x32_bf16(false, a, false, b,
                                                 (short)0, c, false, false);
}

// 64x96 GEMM tile-set for this WG: G[64][96] = A[64][K] * Wp[96][K]^T
// wave w: N-tile = w%6, M-pair = w/6 (rows [32*mp, 32*mp+32)).
__device__ __forceinline__ void gemm96(const __bf16* __restrict__ A, int K,
                                       const __bf16* __restrict__ Wp,
                                       float* __restrict__ G) {
  const int tid  = threadIdx.x;
  const int lane = tid & 31;
  const int wave = tid >> 5;
  const int nt   = wave % 6;
  const int mp   = wave / 6;
  const int lrow = lane & 15;
  const int half = lane >> 4;

  const __bf16* wrow = Wp + (nt * 16 + lrow) * K;   // B fragment row (global, L2)
  const __bf16* ar0  = A + (mp * 32 + lrow) * K;    // A fragment rows (LDS)
  const __bf16* ar1  = ar0 + 16 * K;

  v8f acc0 = {};
  v8f acc1 = {};
#pragma unroll 4
  for (int k0 = 0; k0 < K; k0 += 32) {
    v16bf bf = load_frag(wrow + k0, half);
    v16bf a0 = load_frag(ar0 + k0, half);
    v16bf a1 = load_frag(ar1 + k0, half);
    acc0 = wmma_bf16(a0, bf, acc0);
    acc1 = wmma_bf16(a1, bf, acc1);
  }

  // D layout: vgpr r -> row m0+r, lane -> col; lanes 16..31 are rows +8.
  const int col = nt * 16 + lrow;
  const int m0  = mp * 32 + half * 8;
#pragma unroll
  for (int r = 0; r < 8; ++r) {
    G[(m0 + r) * COLS_ + col]      = acc0[r];
    G[(m0 + 16 + r) * COLS_ + col] = acc1[r];
  }
}

// Cluster barrier (HW, ~100ns on one SE) + monotonic atomic fallback barrier.
__device__ __forceinline__ void gbar(unsigned* bar, int target) {
  __threadfence();
  __syncthreads();
  if (threadIdx.x == 0) {
    asm volatile("s_barrier_signal -3" ::: "memory");   // cluster user barrier
    __hip_atomic_fetch_add(bar, 1u, __ATOMIC_RELEASE, __HIP_MEMORY_SCOPE_AGENT);
  }
  asm volatile("s_barrier_wait -3" ::: "memory");
  if (threadIdx.x == 0) {
    while (__hip_atomic_load(bar, __ATOMIC_ACQUIRE, __HIP_MEMORY_SCOPE_AGENT) <
           (unsigned)target)
      __builtin_amdgcn_s_sleep(2);
  }
  __syncthreads();
  __threadfence();
}

#define LNP(L, XH, GB, CC) ((((L) * 2 + (XH)) * 2 + (GB)) * COLS_ + (CC))

__device__ __forceinline__ float sigm(float x) { return 1.f / (1.f + __expf(-x)); }

__global__ __launch_bounds__(NTHR_, 1) void gru_persistent(
    const float* __restrict__ xs,
    const __bf16* __restrict__ wx0p, const __bf16* __restrict__ wh0p,
    const __bf16* __restrict__ wx1p, const __bf16* __restrict__ wh1p,
    const float* __restrict__ gx0, const float* __restrict__ bx0,
    const float* __restrict__ gh0, const float* __restrict__ bh0,
    const float* __restrict__ gx1, const float* __restrict__ bx1,
    const float* __restrict__ gh1, const float* __restrict__ bh1,
    __bf16* __restrict__ h0g, __bf16* __restrict__ h1g,
    float* __restrict__ S, unsigned* __restrict__ bar,
    float* __restrict__ out) {
  extern __shared__ char smem[];
  __bf16* xbuf  = (__bf16*)(smem);                // [64][256] bf16 : 32 KB
  __bf16* h0buf = (__bf16*)(smem + 32768);        // [64][512] bf16 : 64 KB
  __bf16* h1buf = (__bf16*)(smem + 98304);        // [64][512] bf16 : 64 KB
  float*  gxbuf = (float*)(smem + 163840);        // [64][96]  f32  : 24 KB
  float*  ghbuf = (float*)(smem + 188416);        // [64][96]  f32  : 24 KB
  float*  lnp   = (float*)(smem + 212992);        // [2][2][2][96]  :  3 KB

  const int tid = threadIdx.x;
  const int wg  = blockIdx.x;

  // ---- one-time: preload LN gamma/beta for this WG's 96 gate columns ------
  for (int cc = tid; cc < COLS_; cc += NTHR_) {
    const int g = ((cc >> 5) << 9) + (wg << 5) + (cc & 31); // chunk*512+wg*32+c
    lnp[LNP(0, 0, 0, cc)] = gx0[g];  lnp[LNP(0, 0, 1, cc)] = bx0[g];
    lnp[LNP(0, 1, 0, cc)] = gh0[g];  lnp[LNP(0, 1, 1, cc)] = bh0[g];
    lnp[LNP(1, 0, 0, cc)] = gx1[g];  lnp[LNP(1, 0, 1, cc)] = bx1[g];
    lnp[LNP(1, 1, 0, cc)] = gh1[g];  lnp[LNP(1, 1, 1, cc)] = bh1[g];
  }
  // initial states (h0g/h1g pre-zeroed) and x_0
  for (int e8 = tid; e8 < (B_ * H_) / 8; e8 += NTHR_) {
    ((uint4*)h0buf)[e8] = ((const uint4*)h0g)[e8];
    ((uint4*)h1buf)[e8] = ((const uint4*)h1g)[e8];
  }
  for (int e = tid; e < B_ * CIN_; e += NTHR_)
    xbuf[e] = (__bf16)xs[(e << 10) + 0];          // xs[(b*256+c)*1024 + t]
  __syncthreads();

  const __bf16* wx0s = wx0p + wg * COLS_ * CIN_;
  const __bf16* wh0s = wh0p + wg * COLS_ * H_;
  const __bf16* wx1s = wx1p + wg * COLS_ * H_;
  const __bf16* wh1s = wh1p + wg * COLS_ * H_;

  int nbar = 0;
  for (int t = 0; t < T_; ++t) {
#pragma unroll 1
    for (int layer = 0; layer < 2; ++layer) {
      // ---- GEMMs into LDS gate buffers ----
      if (layer == 0) {
        gemm96(xbuf, CIN_, wx0s, gxbuf);
        gemm96(h0buf, H_, wh0s, ghbuf);
      } else {
        gemm96(h0buf, H_, wx1s, gxbuf);
        gemm96(h1buf, H_, wh1s, ghbuf);
      }
      __syncthreads();

      // ---- partial LN stats over this WG's 96 columns -> global atomics ----
      if (tid < B_) {
        float sx = 0.f, sxx = 0.f, sh = 0.f, shh = 0.f;
        const float* gx = gxbuf + tid * COLS_;
        const float* gh = ghbuf + tid * COLS_;
#pragma unroll 4
        for (int c = 0; c < COLS_; ++c) {
          float a = gx[c], d = gh[c];
          sx += a; sxx += a * a; sh += d; shh += d * d;
        }
        float* s = S + (layer * B_ + tid) * 4;
        atomicAdd(s + 0, sx);  atomicAdd(s + 1, sxx);
        atomicAdd(s + 2, sh);  atomicAdd(s + 3, shh);
      }
      gbar(bar, ++nbar);

      // ---- normalize + GRU combine for this WG's 32 hidden columns ----
      const __bf16* holdb = (layer == 0) ? h0buf : h1buf;
      __bf16* hg = (layer == 0) ? h0g : h1g;
      const float i3h = 1.0f / (float)G3_;
      for (int e = tid; e < B_ * HC_; e += NTHR_) {
        const int b = e >> 5, c = e & 31;
        const float* s = S + (layer * B_ + b) * 4;
        float mx = s[0] * i3h, vx = fmaxf(s[1] * i3h - mx * mx, 0.f);
        float mh = s[2] * i3h, vh = fmaxf(s[3] * i3h - mh * mh, 0.f);
        float rsx = rsqrtf(vx + EPS_), rsh = rsqrtf(vh + EPS_);
        const int c0 = c, c1 = c + 32, c2 = c + 64;
        const float* gb = gxbuf + b * COLS_;
        const float* hb = ghbuf + b * COLS_;
        float ir = (gb[c0] - mx) * rsx * lnp[LNP(layer,0,0,c0)] + lnp[LNP(layer,0,1,c0)];
        float ii = (gb[c1] - mx) * rsx * lnp[LNP(layer,0,0,c1)] + lnp[LNP(layer,0,1,c1)];
        float in_= (gb[c2] - mx) * rsx * lnp[LNP(layer,0,0,c2)] + lnp[LNP(layer,0,1,c2)];
        float hr = (hb[c0] - mh) * rsh * lnp[LNP(layer,1,0,c0)] + lnp[LNP(layer,1,1,c0)];
        float hi = (hb[c1] - mh) * rsh * lnp[LNP(layer,1,0,c1)] + lnp[LNP(layer,1,1,c1)];
        float hn = (hb[c2] - mh) * rsh * lnp[LNP(layer,1,0,c2)] + lnp[LNP(layer,1,1,c2)];
        float r  = sigm(ir + hr);
        float ig = sigm(ii + hi);
        float n  = tanhf(in_ + r * hn);
        const int hcol = wg * HC_ + c;
        float hold = (float)holdb[b * H_ + hcol];
        float hnew = n + ig * (hold - n);
        hg[b * H_ + hcol] = (__bf16)hnew;
        if (layer == 1) out[(b * H_ + hcol) * T_ + t] = hnew;
      }
      gbar(bar, ++nbar);

      // ---- reload full new state into LDS; reset this layer's stats ----
      const __bf16* src = (layer == 0) ? h0g : h1g;
      __bf16* dst = (layer == 0) ? h0buf : h1buf;
      for (int e8 = tid; e8 < (B_ * H_) / 8; e8 += NTHR_)
        ((uint4*)dst)[e8] = ((const uint4*)src)[e8];
      for (int e = tid; e < B_ * 4; e += NTHR_)   // benign identical-zero race
        S[layer * B_ * 4 + e] = 0.f;
      if (layer == 1 && t + 1 < T_) {
        for (int e = tid; e < B_ * CIN_; e += NTHR_)
          xbuf[e] = (__bf16)xs[(e << 10) + (t + 1)];
      }
      __syncthreads();
    }
  }
}

// ---------------------------------------------------------------------------
// Weight pack: packed row p = wg*96 + cc maps to original gate row
// g = (cc/32)*512 + wg*32 + (cc%32); K contiguous, bf16 => B-fragment rows are
// two 16-byte contiguous chunks per lane per K-slab.
// ---------------------------------------------------------------------------
__global__ void pack_w(const float* __restrict__ W, __bf16* __restrict__ dst,
                       int K, int total) {
  int i = blockIdx.x * 256 + threadIdx.x;
  if (i >= total) return;
  int k = i % K;
  int p = i / K;
  int j = p / COLS_;
  int cc = p - j * COLS_;
  int g = ((cc >> 5) << 9) + (j << 5) + (cc & 31);
  dst[i] = (__bf16)W[g * K + k];
}

__global__ void init_ws(__bf16* h0g, __bf16* h1g, float* S, unsigned* bar) {
  int i = blockIdx.x * 256 + threadIdx.x;
  if (i < B_ * H_) { h0g[i] = (__bf16)0.f; h1g[i] = (__bf16)0.f; }
  if (i < 2 * B_ * 4) S[i] = 0.f;
  if (i == 0) *bar = 0u;
}

extern "C" void kernel_launch(void* const* d_in, const int* in_sizes, int n_in,
                              void* d_out, int out_size, void* d_ws,
                              size_t ws_size, hipStream_t stream) {
  (void)in_sizes; (void)n_in; (void)out_size; (void)ws_size;
  const float* xs  = (const float*)d_in[0];
  const float* Wx0 = (const float*)d_in[1];
  const float* Wh0 = (const float*)d_in[2];
  const float* gx0 = (const float*)d_in[3];
  const float* bx0 = (const float*)d_in[4];
  const float* gh0 = (const float*)d_in[5];
  const float* bh0 = (const float*)d_in[6];
  const float* Wx1 = (const float*)d_in[7];
  const float* Wh1 = (const float*)d_in[8];
  const float* gx1 = (const float*)d_in[9];
  const float* bx1 = (const float*)d_in[10];
  const float* gh1 = (const float*)d_in[11];
  const float* bh1 = (const float*)d_in[12];
  float* out = (float*)d_out;

  char* ws = (char*)d_ws;
  __bf16* wx0p = (__bf16*)(ws + 0);              //  786432 B
  __bf16* wh0p = (__bf16*)(ws + 786432);         // 1572864 B
  __bf16* wx1p = (__bf16*)(ws + 2359296);        // 1572864 B
  __bf16* wh1p = (__bf16*)(ws + 3932160);        // 1572864 B
  __bf16* h0g  = (__bf16*)(ws + 5505024);        //   65536 B
  __bf16* h1g  = (__bf16*)(ws + 5570560);        //   65536 B
  float*  S    = (float*)(ws + 5636096);         //    2048 B
  unsigned* bar = (unsigned*)(ws + 5638144);     //       4 B

  const int nx = NWG_ * COLS_ * CIN_;            // 393216
  const int nh = NWG_ * COLS_ * H_;              // 786432
  pack_w<<<(nx + 255) / 256, 256, 0, stream>>>(Wx0, wx0p, CIN_, nx);
  pack_w<<<(nh + 255) / 256, 256, 0, stream>>>(Wh0, wh0p, H_, nh);
  pack_w<<<(nh + 255) / 256, 256, 0, stream>>>(Wx1, wx1p, H_, nh);
  pack_w<<<(nh + 255) / 256, 256, 0, stream>>>(Wh1, wh1p, H_, nh);
  init_ws<<<128, 256, 0, stream>>>(h0g, h1g, S, bar);

  // One 16-WG cluster (one WGP each), 12 waves/WG, ~211 KB dynamic LDS.
  gru_persistent<<<NWG_, NTHR_, 216064, stream>>>(
      xs, wx0p, wh0p, wx1p, wh1p,
      gx0, bx0, gh0, bh0, gx1, bx1, gh1, bh1,
      h0g, h1g, S, bar, out);
}
